// MMD_GCNNet_57904749084743
// MI455X (gfx1250) — compile-verified, matrix operand order checked by reading, not run on hardware
//
#include <hip/hip_runtime.h>
#include <math.h>

typedef __attribute__((ext_vector_type(16))) _Float16 v16h;
typedef __attribute__((ext_vector_type(8)))  float    v8f;

#define NN    10000
#define DEG   16
#define FIN   512
#define HID   16
#define NC    32
#define DNB   17                 // neighbors incl. self-loop
#define E0    (NN*DEG)           // 160000 real edges
#define ETOT  (NN*(DEG+1))       // 170000 with self-loops
#define K2E   14.426950408889634f   // 10 * log2(e)  (sigma = 0.1)
#define PADBIG 1e30f

__device__ __forceinline__ int gwid() { return (int)((blockIdx.x * blockDim.x + threadIdx.x) >> 5); }
__device__ __forceinline__ int lid()  { return (int)(threadIdx.x & 31); }

// K index within a 32-chunk for fragment slot j of this lane (16-bit A/B layout):
// lanes 0-15 hold K {0..7,16..23}; lanes 16-31 hold K {8..15,24..31}
__device__ __forceinline__ int kslot(int j, int hi) { return ((j < 8) ? j : j + 8) + hi * 8; }

// ---------------- GEMM1: t1 = x @ W1 + b1   [N,512]x[512,16] ----------------
__global__ void gemm1_wmma(const float* __restrict__ x, const float* __restrict__ W1,
                           const float* __restrict__ b1, float* __restrict__ t1) {
    int w = gwid(); if (w >= NN / 16) return;
    int lane = lid(), lo = lane & 15, hi = lane >> 4;
    int row = w * 16 + lo;
    v8f acc = {};
    for (int kb = 0; kb < FIN / 32; ++kb) {
        v16h a, b;
#pragma unroll
        for (int j = 0; j < 16; ++j) {
            int k = kb * 32 + kslot(j, hi);
            a[j] = (_Float16)x[row * FIN + k];
            b[j] = (_Float16)W1[k * HID + lo];     // B[k][n], n = lo
        }
        acc = __builtin_amdgcn_wmma_f32_16x16x32_f16(false, a, false, b, (short)0, acc, false, false);
    }
#pragma unroll
    for (int v = 0; v < 8; ++v) {
        int m = v + 8 * hi;
        t1[(w * 16 + m) * HID + lo] = acc[v] + b1[lo];
    }
}

// ---------------- GEMM2: t2 = h1 @ W2 + b2   [N,16]x[16,32], K padded to 32 ----------------
__global__ void gemm2_wmma(const float* __restrict__ h1, const float* __restrict__ W2,
                           const float* __restrict__ b2, float* __restrict__ t2) {
    int w = gwid(); if (w >= (NN / 16) * 2) return;
    int rt = w >> 1, ct = w & 1;
    int lane = lid(), lo = lane & 15, hi = lane >> 4;
    int row = rt * 16 + lo, n = ct * 16 + lo;
    v16h a, b;
#pragma unroll
    for (int j = 0; j < 16; ++j) {
        int k = kslot(j, hi);
        a[j] = (k < HID) ? (_Float16)h1[row * HID + k] : (_Float16)0.f;
        b[j] = (k < HID) ? (_Float16)W2[k * NC + n]    : (_Float16)0.f;
    }
    v8f acc = {};
    acc = __builtin_amdgcn_wmma_f32_16x16x32_f16(false, a, false, b, (short)0, acc, false, false);
#pragma unroll
    for (int v = 0; v < 8; ++v) {
        int m = v + 8 * hi;
        t2[(rt * 16 + m) * NC + n] = acc[v] + b2[n];
    }
}

// ---- Build per-node neighbor matrix (17 x H, padded 32 x 32) in WMMA fragment layout ----
// frag[node][tile][lane][16] f16 ; tile rows d = tile*16 + (lane%16); same image is A for
// the row node and B for the col node of a Gram product.
__global__ void gather_frag(const float* __restrict__ t, const int* __restrict__ row0,
                            _Float16* __restrict__ frag, int H) {
    int w = gwid(); if (w >= NN * 2) return;
    int node = w >> 1, tile = w & 1;
    int lane = lid(), lo = lane & 15, hi = lane >> 4;
    int d = tile * 16 + lo;
    int src = -1;
    if (d < DEG)       src = row0[node * DEG + d];
    else if (d == DEG) src = node;                 // self-loop row
    _Float16* out = frag + ((size_t)w * 32 + (size_t)lane) * 16;
#pragma unroll
    for (int j = 0; j < 16; ++j) {
        int k = kslot(j, hi);
        float v = (src >= 0 && k < H) ? t[src * H + k] : 0.f;
        out[j] = (_Float16)v;
    }
}

// ---- sq[node][d] = ||h_d||^2 (through f16 rounding, consistent with WMMA inputs); pad=1e30 ----
__global__ void compute_sq(const float* __restrict__ t, const int* __restrict__ row0,
                           float* __restrict__ sqpad, int H) {
    int idx = (int)(blockIdx.x * blockDim.x + threadIdx.x);
    if (idx >= NN * 32) return;
    int node = idx >> 5, d = idx & 31;
    float s;
    if (d < DNB) {
        int src = (d < DEG) ? row0[node * DEG + d] : node;
        s = 0.f;
        for (int k = 0; k < H; ++k) {
            float v = (float)(_Float16)t[src * H + k];
            s += v * v;
        }
    } else {
        s = PADBIG;   // padded rows -> exp2(-huge) == 0, branch-free masking
    }
    sqpad[idx] = s;
}

// ---- mean_{a,b<17} exp(-(sq_a+sq_b-2*dot)/sigma) via 4 WMMAs (32x32 padded Gram) ----
__device__ float gram_mean(const v16h* __restrict__ frag, const float* __restrict__ sqpad,
                           int r, int c, int lane) {
    int lo = lane & 15, hi = lane >> 4;
    float sr = sqpad[r * 32 + lane];
    float sc = sqpad[c * 32 + lane];
    float acc = 0.f;
#pragma unroll
    for (int rt = 0; rt < 2; ++rt) {
        v16h a = frag[((size_t)r * 2 + rt) * 32 + lane];
#pragma unroll
        for (int ct = 0; ct < 2; ++ct) {
            v16h b = frag[((size_t)c * 2 + ct) * 32 + lane];
            v8f g = {};
            g = __builtin_amdgcn_wmma_f32_16x16x32_f16(false, a, false, b, (short)0, g, false, false);
            float sqb = __shfl(sc, ct * 16 + lo, 32);   // col index b = ct*16 + n
#pragma unroll
            for (int v = 0; v < 8; ++v) {
                int aidx = rt * 16 + v + 8 * hi;        // row index a
                float sqa = __shfl(sr, aidx, 32);
                acc += exp2f((2.f * g[v] - sqa - sqb) * K2E);
            }
        }
    }
#pragma unroll
    for (int off = 16; off > 0; off >>= 1)
        acc += __shfl_xor(acc, off, 32);
    return acc * (1.f / (float)(DNB * DNB));
}

__global__ void selfmean_kernel(const v16h* __restrict__ frag, const float* __restrict__ sqpad,
                                float* __restrict__ selfm) {
    int w = gwid(); if (w >= NN) return;
    float m = gram_mean(frag, sqpad, w, w, lid());
    if (lid() == 0) selfm[w] = m;
}

// one wave per edge; self-loop edges (e >= E0) are exactly 0
__global__ void edge_mmd_kernel(const v16h* __restrict__ frag, const float* __restrict__ sqpad,
                                const float* __restrict__ selfm,
                                const int* __restrict__ row0, const int* __restrict__ col0,
                                float* __restrict__ mmd_out) {
    int w = gwid(); if (w >= ETOT) return;
    if (w >= E0) { if (lid() == 0) mmd_out[w] = 0.f; return; }
    int r = row0[w], c = col0[w];
    float cm = gram_mean(frag, sqpad, r, c, lid());
    if (lid() == 0) mmd_out[w] = selfm[r] + selfm[c] - 2.f * cm;
}

// ---- GCN aggregation: constant norm 1/17 (in-degree is 17 everywhere with self-loops) ----
__global__ void aggregate(const float* __restrict__ t, const int* __restrict__ row0,
                          float* __restrict__ out, int H, int do_relu) {
    int idx = (int)(blockIdx.x * blockDim.x + threadIdx.x);
    if (idx >= NN * H) return;
    int node = idx / H, f = idx - node * H;
    float s = t[node * H + f];                      // self loop
#pragma unroll 4
    for (int d = 0; d < DEG; ++d)
        s += t[row0[node * DEG + d] * H + f];
    s *= (1.f / (float)DNB);
    if (do_relu) s = fmaxf(s, 0.f);
    out[idx] = s;
}

// ---- log-softmax over 32 classes == one wave32 per node ----
__global__ void logsoftmax_kernel(const float* __restrict__ h2, float* __restrict__ out) {
    int w = gwid(); if (w >= NN) return;
    int lane = lid();
    float v = h2[w * NC + lane];
    float m = v;
#pragma unroll
    for (int off = 16; off > 0; off >>= 1) m = fmaxf(m, __shfl_xor(m, off, 32));
    float e = __expf(v - m);
    float s = e;
#pragma unroll
    for (int off = 16; off > 0; off >>= 1) s += __shfl_xor(s, off, 32);
    out[w * NC + lane] = v - m - __logf(s);
}

extern "C" void kernel_launch(void* const* d_in, const int* in_sizes, int n_in,
                              void* d_out, int out_size, void* d_ws, size_t ws_size,
                              hipStream_t stream) {
    const float* x  = (const float*)d_in[0];
    const int*   ei = (const int*)  d_in[1];
    const float* W1 = (const float*)d_in[2];
    const float* b1 = (const float*)d_in[3];
    const float* W2 = (const float*)d_in[4];
    const float* b2 = (const float*)d_in[5];
    const int* row0 = ei;
    const int* col0 = ei + E0;
    float* out = (float*)d_out;

    // workspace carve-up (~24.4 MB)
    char*  ws    = (char*)d_ws;
    float* tbuf  = (float*)ws;            // N*32 f32 : t1 (uses N*16) / t2
    float* hbuf  = tbuf + NN * NC;        // N*32 f32 : h1 / h2
    float* sqpad = hbuf + NN * NC;        // N*32 f32
    float* selfm = sqpad + NN * 32;       // N f32
    size_t off = (size_t)(NN * NC * 2 + NN * 32 + NN) * sizeof(float);
    off = (off + 255) & ~(size_t)255;     // 32B-align v16h fragment buffer
    _Float16* frag = (_Float16*)(ws + off);           // N*2*32*16 f16 = 20.48 MB
    const v16h* fragv = (const v16h*)frag;

    const int BLK = 256;                  // 8 waves per block
    const int gGemm1 = (625 * 32 + BLK - 1) / BLK;
    const int gGemm2 = (1250 * 32 + BLK - 1) / BLK;
    const int gGath  = (NN * 2 * 32 + BLK - 1) / BLK;
    const int gSq    = (NN * 32 + BLK - 1) / BLK;
    const int gSelf  = (NN * 32 + BLK - 1) / BLK;     // NN waves
    const int gEdge  = (ETOT * 32 + BLK - 1) / BLK;
    const int gLSM   = (NN * 32 + BLK - 1) / BLK;     // NN waves

    // ================= Layer 1 (H = 16) =================
    gemm1_wmma<<<gGemm1, BLK, 0, stream>>>(x, W1, b1, tbuf);
    gather_frag<<<gGath, BLK, 0, stream>>>(tbuf, row0, frag, HID);
    compute_sq<<<gSq, BLK, 0, stream>>>(tbuf, row0, sqpad, HID);
    selfmean_kernel<<<gSelf, BLK, 0, stream>>>(fragv, sqpad, selfm);
    edge_mmd_kernel<<<gEdge, BLK, 0, stream>>>(fragv, sqpad, selfm, row0, col0,
                                               out + (size_t)NN * NC);
    {
        const int gAgg = (NN * HID + BLK - 1) / BLK;
        aggregate<<<gAgg, BLK, 0, stream>>>(tbuf, row0, hbuf, HID, /*relu=*/1);
    }

    // ================= Layer 2 (H = 32) =================
    gemm2_wmma<<<gGemm2, BLK, 0, stream>>>(hbuf, W2, b2, tbuf);
    gather_frag<<<gGath, BLK, 0, stream>>>(tbuf, row0, frag, NC);
    compute_sq<<<gSq, BLK, 0, stream>>>(tbuf, row0, sqpad, NC);
    selfmean_kernel<<<gSelf, BLK, 0, stream>>>(fragv, sqpad, selfm);
    edge_mmd_kernel<<<gEdge, BLK, 0, stream>>>(fragv, sqpad, selfm, row0, col0,
                                               out + (size_t)NN * NC + ETOT);
    {
        const int gAgg = (NN * NC + BLK - 1) / BLK;
        aggregate<<<gAgg, BLK, 0, stream>>>(tbuf, row0, hbuf, NC, /*relu=*/0);
    }
    logsoftmax_kernel<<<gLSM, BLK, 0, stream>>>(hbuf, out);
}
